// spatialAttention_67061619359902
// MI455X (gfx1250) — matrix operand order repaired and checked by reading, not compiled, not run on hardware
//
#include <hip/hip_runtime.h>
#include <hip/hip_bf16.h>
#include <math.h>

typedef __attribute__((ext_vector_type(16))) _Float16 v16h;
typedef __attribute__((ext_vector_type(8)))  _Float16 v8h;
typedef __attribute__((ext_vector_type(4)))  _Float16 v4h;
typedef __attribute__((ext_vector_type(8)))  float    v8f;

#define N_V   128   // vessels (i and j)
#define H_DIM 128   // hidden dim
#define LDW   136   // padded LDS stride in f16 (272 B, 16B-aligned rows)
#define MROWS 64    // rows of W handled per block (M-half)

#define LOG2E    1.4426950408889634f
#define LN2      0.6931471805599453f
#define TWOLOG2E 2.8853900817779268f

__device__ __forceinline__ float fast_softplus(float x) {
    // softplus(x) = max(x,0) + ln2 * log2(1 + 2^(-|x|*log2e))
    // branch-free; x = -1e24 -> exp2 underflows to 0 -> result 0 exactly
    float e = __builtin_amdgcn_exp2f(-fabsf(x) * LOG2E);
    return fmaxf(x, 0.0f) + LN2 * __builtin_amdgcn_logf(1.0f + e);
}

__device__ __forceinline__ float fast_tanh(float x) {
#if __has_builtin(__builtin_amdgcn_tanhf)
    return __builtin_amdgcn_tanhf(x);   // v_tanh_f32
#else
    float e = __builtin_amdgcn_exp2f(-TWOLOG2E * fabsf(x));   // 2^(-2|x|log2e)
    float t = (1.0f - e) * __builtin_amdgcn_rcpf(1.0f + e);
    return __builtin_copysignf(t, x);
#endif
}

__global__ __launch_bounds__(256) void spatial_attn_wmma(
    const float* __restrict__ hidden,        // (N, B, H)
    const float* __restrict__ dist,          // (B, N, N)
    const float* __restrict__ bear,          // (B, N, N)
    const float* __restrict__ head,          // (B, N, N)
    const unsigned char* __restrict__ smask, // (B, N) bool
    const float* __restrict__ domain,        // (24, 24)
    float* __restrict__ out,                 // (B, N, H)
    int B)
{
    __shared__ __align__(16) _Float16 sW[MROWS * LDW];  // A: W rows [i_local][j]
    __shared__ __align__(16) _Float16 sH[N_V * LDW];    // B: H^T   [h][j]
    __shared__ float sDom[24 * 24];
    __shared__ unsigned char sMask[N_V];

    const int tid = threadIdx.x;
    const int b   = blockIdx.x;
    const int mh  = blockIdx.y;  // which 64-row half of i

    // ---- Phase 0: stage domain table + sequence mask ----
    for (int idx = tid; idx < 24 * 24; idx += 256) sDom[idx] = domain[idx];
    if (tid < N_V) sMask[tid] = smask[b * N_V + tid];

    // ---- Phase 1: stage H^T as f16: sH[h][j] = hidden[j, b, h] ----
    // vectorized float4 over h (contiguous global dim); scalar transpose into LDS
    {
        const float4* hrow = (const float4*)(hidden + (size_t)b * H_DIM);
        #pragma unroll 4
        for (int it = 0; it < (N_V * H_DIM / 4) / 256; ++it) {
            int idx = it * 256 + tid;         // 0..4095
            int j  = idx >> 5;                // 0..127
            int hv = idx & 31;                // float4 index along h
            float4 v = hrow[(size_t)j * (B * H_DIM / 4) + hv];
            int h = hv * 4;
            sH[(h + 0) * LDW + j] = (_Float16)v.x;
            sH[(h + 1) * LDW + j] = (_Float16)v.y;
            sH[(h + 2) * LDW + j] = (_Float16)v.z;
            sH[(h + 3) * LDW + j] = (_Float16)v.w;
        }
    }

    __syncthreads();

    // ---- Phase 2: fused weight construction: sW[li][j] = softplus(dom - dist) ----
    // vectorized float4 over j
    {
        const size_t base = (size_t)b * N_V * N_V + (size_t)mh * MROWS * N_V;
        const float4* d4 = (const float4*)(dist + base);
        const float4* b4 = (const float4*)(bear + base);
        const float4* h4 = (const float4*)(head + base);
        #pragma unroll 2
        for (int it = 0; it < (MROWS * N_V / 4) / 256; ++it) {
            int idx = it * 256 + tid;          // 0..2047
            int li = idx >> 5;                 // 0..63
            int jv = idx & 31;                 // float4 index along j
            int i  = mh * MROWS + li;
            float mi = (sMask[i] != 0) ? 0.0f : 1.0f;  // 1 => force mask fill
            float4 dm = d4[idx];
            float4 hm = h4[idx];
            float4 bm = b4[idx];
            int j0 = jv * 4;
            v4h wpk;
            #pragma unroll
            for (int c = 0; c < 4; ++c) {
                float hmc = (&hm.x)[c];
                float bmc = (&bm.x)[c];
                float dmc = (&dm.x)[c];
                int i1 = (int)floorf(hmc * (1.0f / 15.0f));
                i1 = i1 < 0 ? 0 : (i1 > 23 ? 23 : i1);
                int i2 = (int)floorf(bmc * (1.0f / 15.0f));
                i2 = i2 < 0 ? 0 : (i2 > 23 ? 23 : i2);
                float mj = (sMask[j0 + c] != 0) ? 0.0f : 1.0f;
                float bad = fmaxf(mi, mj);                 // 1 if either masked
                float d = dmc + bad * 1e24f;               // branchless mask fill
                wpk[c] = (_Float16)fast_softplus(sDom[i1 * 24 + i2] - d);
            }
            *(v4h*)&sW[li * LDW + j0] = wpk;
        }
    }

    __syncthreads();

    // ---- Phase 3: 64x128 = (64x128)W x (128x128)H^T via v_wmma_f32_16x16x32_f16 ----
    const int wave = tid >> 5;
    const int lane = tid & 31;
    const int half = lane >> 4;   // 0: lanes 0-15, 1: lanes 16-31
    const int lr   = lane & 15;
    const int mt   = wave >> 1;          // M tile 0..3
    const int nt0  = (wave & 1) * 4;     // N tile base 0 or 4

    // Per-output-row mask selects, hoisted out of the N-tile loop.
    float rowm[8];
    #pragma unroll
    for (int r = 0; r < 8; ++r)
        rowm[r] = (sMask[mh * MROWS + mt * 16 + r + half * 8] != 0) ? 1.0f : 0.0f;

    // A fragments: 16-bit A 16x32 layout — lanes 0-15 row M=lr hold K {0..7,16..23};
    // lanes 16-31 hold K {8..15, 24..31}.
    union { v16h v; v8h h[2]; } afrag[4];
    const int arow = mt * 16 + lr;
    #pragma unroll
    for (int kt = 0; kt < 4; ++kt) {
        int kb = kt * 32 + half * 8;
        afrag[kt].h[0] = *(const v8h*)&sW[arow * LDW + kb];
        afrag[kt].h[1] = *(const v8h*)&sW[arow * LDW + kb + 16];
    }

    #pragma unroll
    for (int t = 0; t < 4; ++t) {
        int nt = nt0 + t;
        int n  = nt * 16 + lr;
        v8f acc = {};
        #pragma unroll
        for (int kt = 0; kt < 4; ++kt) {
            // B 32x16: lane col n; lanes 0-15 hold K {0..7,16..23}, lanes 16-31 K {8..15,24..31};
            // two aligned 16B LDS reads per lane.
            int kb = kt * 32 + half * 8;
            union { v16h v; v8h h[2]; } bu;
            bu.h[0] = *(const v8h*)&sH[n * LDW + kb];
            bu.h[1] = *(const v8h*)&sH[n * LDW + kb + 16];
            v16h bfrag = bu.v;
            acc = __builtin_amdgcn_wmma_f32_16x16x32_f16(
                false, afrag[kt].v, false, bfrag, (short)0, acc, false, false);
        }
        // C/D layout: VGPR r, lane L -> M = r + 8*(L>=16), N = L&15
        #pragma unroll
        for (int r = 0; r < 8; ++r) {
            int i_row = mh * MROWS + mt * 16 + r + half * 8;
            int hcol  = nt * 16 + lr;
            float v = rowm[r] * fast_tanh(acc[r]);   // branchless row mask
            out[(size_t)b * N_V * H_DIM + (size_t)i_row * H_DIM + hcol] = v;
        }
    }
}

extern "C" void kernel_launch(void* const* d_in, const int* in_sizes, int n_in,
                              void* d_out, int out_size, void* d_ws, size_t ws_size,
                              hipStream_t stream) {
    const float*         hidden = (const float*)d_in[0];         // (N,B,H)
    const float*         dist   = (const float*)d_in[1];         // (B,N,N)
    const float*         bear   = (const float*)d_in[2];         // (B,N,N)
    const float*         head   = (const float*)d_in[3];         // (B,N,N)
    const unsigned char* smask  = (const unsigned char*)d_in[4]; // (B,N) bool
    const float*         domain = (const float*)d_in[5];         // (24,24)
    float*               out    = (float*)d_out;                 // (B,N,H)

    int B = in_sizes[0] / (N_V * H_DIM);  // hidden has N*B*H elements

    dim3 grid(B, 2), block(256);
    spatial_attn_wmma<<<grid, block, 0, stream>>>(hidden, dist, bear, head,
                                                  smask, domain, out, B);
}
